// VisualContextFusion_35725537968656
// MI455X (gfx1250) — matrix operand order repaired and checked by reading, not compiled, not run on hardware
//
#include <hip/hip_runtime.h>
#include <math.h>

typedef __attribute__((ext_vector_type(16))) _Float16 v16h;
typedef __attribute__((ext_vector_type(8)))  float    v8f;

#define INF_C 3000.0f
#define KNN   5

// ---------------------------------------------------------------------------
// K1: top-5 nearest neighbors per row from spatial centers.
// One block per batch b, one thread per row i. Centers staged in LDS (4KB).
// Matches reference semantics: padded boxes shifted +INF, self excluded,
// strict '<' insertion => smaller index wins ties (jax top_k tie rule),
// squared distance preserves the sqrt ordering.
// ---------------------------------------------------------------------------
__global__ void __launch_bounds__(512)
topk_kernel(const float* __restrict__ spatials, const int* __restrict__ mask,
            int* __restrict__ nbr, int n) {
  const int b = blockIdx.x;
  const int i = threadIdx.x;
  extern __shared__ float cxy[];  // 2*n floats
  const float* sp = spatials + ((size_t)b * n + i) * 6;
  float cx = (sp[0] + sp[2]) * 0.5f;
  float cy = (sp[1] + sp[3]) * 0.5f;
  if (mask[b * n + i] == 0) { cx += INF_C; cy += INF_C; }
  cxy[2 * i]     = cx;
  cxy[2 * i + 1] = cy;
  __syncthreads();

  float bd[KNN];
  int   bi[KNN];
#pragma unroll
  for (int t = 0; t < KNN; ++t) { bd[t] = 1e30f; bi[t] = -1; }

  for (int j = 0; j < n; ++j) {
    if (j == i) continue;  // self never among k nearest (ref sets it to INF^2)
    float dx = cx - cxy[2 * j];
    float dy = cy - cxy[2 * j + 1];
    float d  = dx * dx + dy * dy;
    if (d < bd[KNN - 1]) {
      bd[KNN - 1] = d; bi[KNN - 1] = j;
#pragma unroll
      for (int t = KNN - 1; t > 0; --t) {
        if (bd[t] < bd[t - 1]) {
          float td = bd[t]; bd[t] = bd[t - 1]; bd[t - 1] = td;
          int   ti = bi[t]; bi[t] = bi[t - 1]; bi[t - 1] = ti;
        }
      }
    }
  }

  int* out = nbr + ((size_t)b * n + i) * KNN;
#pragma unroll
  for (int t = 0; t < KNN; ++t) {
    int idx = bi[t];
    bool ok = (idx >= 0) && (idx != i) && (mask[b * n + idx] != 0);
    out[t] = ok ? idx : -1;  // -1 => column zeroed by RoI mask / self
  }
}

// ---------------------------------------------------------------------------
// K2: per-batch row sum S_b[h] = sum_i encI[b,i,h]. Coalesced across h.
// ---------------------------------------------------------------------------
__global__ void __launch_bounds__(256)
rowsum_kernel(const float* __restrict__ encI, float* __restrict__ S,
              int n, int H) {
  const int b = blockIdx.x;
  for (int h = threadIdx.x; h < H; h += 256) {
    const float* p = encI + (size_t)b * n * H + h;
    float s = 0.f;
    for (int i = 0; i < n; ++i) s += p[(size_t)i * H];
    S[(size_t)b * H + h] = s;
  }
}

// ---------------------------------------------------------------------------
// K3: gathered dot products sim[b,i,s] = <encI[b,i,:], encI[b,nbr[b,i,s],:]>
// via v_wmma_f32_16x16x32_f16. One wave per 16-row tile; A = tile rows,
// B(slot s) = each column's slot-s neighbor row; we keep the 16x16 diagonal.
// Fragment packing follows CDNA5 ISA VGPR layouts exactly.
// ---------------------------------------------------------------------------
__global__ void __launch_bounds__(32)
sim_wmma_kernel(const float* __restrict__ encI, const int* __restrict__ nbr,
                float* __restrict__ sims, int n, int H, int ntiles) {
  const int b    = blockIdx.x / ntiles;
  const int tile = blockIdx.x % ntiles;
  const int i0   = tile * 16;
  const int lane = threadIdx.x;
  const int col  = lane & 15;   // A-row / B-column owned by this lane
  const int grp  = lane >> 4;   // K-half selector

  // neighbor row per column for each slot (clamp invalid to 0; sim unused)
  int nidx[KNN];
#pragma unroll
  for (int s = 0; s < KNN; ++s) {
    int id = nbr[((size_t)b * n + i0 + col) * KNN + s];
    nidx[s] = (id < 0) ? 0 : id;
  }

  const float* batch = encI + (size_t)b * n * H;
  const float* aRow  = batch + (size_t)(i0 + col) * H;

  v8f c[KNN];
  v8f zero = {};
#pragma unroll
  for (int s = 0; s < KNN; ++s) c[s] = zero;

  for (int ch = 0; ch < H; ch += 32) {
    // A fragment (16-bit A 16x32): lane grp 0 -> K {0..7,16..23}+ch,
    //                              lane grp 1 -> K {8..15,24..31}+ch
    const float* ap = aRow + ch + grp * 8;
    float4 a0 = *(const float4*)(ap + 0);
    float4 a1 = *(const float4*)(ap + 4);
    float4 a2 = *(const float4*)(ap + 16);
    float4 a3 = *(const float4*)(ap + 20);
    v16h a;
    a[0]  = (_Float16)a0.x; a[1]  = (_Float16)a0.y; a[2]  = (_Float16)a0.z; a[3]  = (_Float16)a0.w;
    a[4]  = (_Float16)a1.x; a[5]  = (_Float16)a1.y; a[6]  = (_Float16)a1.z; a[7]  = (_Float16)a1.w;
    a[8]  = (_Float16)a2.x; a[9]  = (_Float16)a2.y; a[10] = (_Float16)a2.z; a[11] = (_Float16)a2.w;
    a[12] = (_Float16)a3.x; a[13] = (_Float16)a3.y; a[14] = (_Float16)a3.z; a[15] = (_Float16)a3.w;

#pragma unroll
    for (int s = 0; s < KNN; ++s) {
      // B fragment (16-bit B 32x16): lane holds column `col`,
      // grp 0 -> K {0..7,16..23}, grp 1 -> K {8..15,24..31} (same as A)
      const float* bp = batch + (size_t)nidx[s] * H + ch + grp * 8;
      float4 b0 = *(const float4*)(bp + 0);
      float4 b1 = *(const float4*)(bp + 4);
      float4 b2 = *(const float4*)(bp + 16);
      float4 b3 = *(const float4*)(bp + 20);
      v16h bm;
      bm[0]  = (_Float16)b0.x; bm[1]  = (_Float16)b0.y; bm[2]  = (_Float16)b0.z; bm[3]  = (_Float16)b0.w;
      bm[4]  = (_Float16)b1.x; bm[5]  = (_Float16)b1.y; bm[6]  = (_Float16)b1.z; bm[7]  = (_Float16)b1.w;
      bm[8]  = (_Float16)b2.x; bm[9]  = (_Float16)b2.y; bm[10] = (_Float16)b2.z; bm[11] = (_Float16)b2.w;
      bm[12] = (_Float16)b3.x; bm[13] = (_Float16)b3.y; bm[14] = (_Float16)b3.z; bm[15] = (_Float16)b3.w;

      c[s] = __builtin_amdgcn_wmma_f32_16x16x32_f16(
          /*neg_a=*/false, a, /*neg_b=*/false, bm,
          /*c_mod=*/(short)0, c[s], /*reuse_a=*/false, /*reuse_b=*/false);
    }
  }

  // Diagonal extraction: C layout => diag d<8 at (vgpr d, lane d);
  // diag d>=8 at (vgpr d-8, lane 16+d) i.e. lane d+16 in 24..31.
#pragma unroll
  for (int s = 0; s < KNN; ++s) {
#pragma unroll
    for (int d = 0; d < 8; ++d) {
      if (lane == d)
        sims[((size_t)b * n + i0 + d) * KNN + s] = c[s][d];
      if (lane == d + 24)
        sims[((size_t)b * n + i0 + d + 8) * KNN + s] = c[s][d];
    }
  }
}

// ---------------------------------------------------------------------------
// K4: epilogue. Closed-form softmax over the sparse row:
//   D = (n-m)*e0 + sum_t e_t,  e0 = exp(-M), e_t = exp(sim_t - M)
//   ctx = (e0*S_b + sum_t (e_t - e0)*encI[nbr_t]) / D
//   out = [encI | ctx]
// One block per row, float4 coalesced.
// ---------------------------------------------------------------------------
__global__ void __launch_bounds__(192)
output_kernel(const float* __restrict__ encI, const int* __restrict__ nbr,
              const float* __restrict__ sims, const float* __restrict__ S,
              float* __restrict__ out, int n, int H) {
  const int row = blockIdx.x;  // b*n + i
  const int b   = row / n;

  int   nid[KNN];
  float sv[KNN];
  int   m  = 0;
  float mx = 0.f;  // row max includes the implicit zeros
#pragma unroll
  for (int t = 0; t < KNN; ++t) {
    nid[t] = nbr[(size_t)row * KNN + t];
    sv[t]  = sims[(size_t)row * KNN + t];
    if (nid[t] >= 0) { ++m; mx = fmaxf(mx, sv[t]); }
  }
  float e0    = __expf(-mx);
  float denom = (float)(n - m) * e0;
  float coef[KNN];
#pragma unroll
  for (int t = 0; t < KNN; ++t) {
    if (nid[t] >= 0) {
      float e = __expf(sv[t] - mx);
      denom  += e;
      coef[t] = e - e0;
    } else {
      coef[t] = 0.f;
    }
  }
  float inv = 1.0f / denom;

  const float* encRow = encI + (size_t)row * H;
  const float* Sb     = S + (size_t)b * H;
  float* o1 = out + (size_t)row * 2 * H;
  float* o2 = o1 + H;

  for (int h = threadIdx.x * 4; h < H; h += blockDim.x * 4) {
    float4 e   = *(const float4*)(encRow + h);
    float4 s4  = *(const float4*)(Sb + h);
    float ax = e0 * s4.x, ay = e0 * s4.y, az = e0 * s4.z, aw = e0 * s4.w;
#pragma unroll
    for (int t = 0; t < KNN; ++t) {
      if (nid[t] >= 0) {
        float4 v = *(const float4*)(encI + ((size_t)b * n + nid[t]) * H + h);
        ax = fmaf(coef[t], v.x, ax);
        ay = fmaf(coef[t], v.y, ay);
        az = fmaf(coef[t], v.z, az);
        aw = fmaf(coef[t], v.w, aw);
      }
    }
    float4 ctx = make_float4(ax * inv, ay * inv, az * inv, aw * inv);
    *(float4*)(o1 + h) = e;
    *(float4*)(o2 + h) = ctx;
  }
}

// ---------------------------------------------------------------------------
extern "C" void kernel_launch(void* const* d_in, const int* in_sizes, int n_in,
                              void* d_out, int out_size, void* d_ws, size_t ws_size,
                              hipStream_t stream) {
  const float* encI     = (const float*)d_in[0];
  const int*   RoI_mask = (const int*)d_in[1];
  const float* spatials = (const float*)d_in[2];
  // d_in[3] is k==5 (compile-time KNN)

  const int BN = in_sizes[1];           // B*n = 32*512
  const int H  = in_sizes[0] / BN;      // 768
  const int n  = 512;
  const int B  = BN / n;

  // workspace: nbr (BN*5 int) | sims (BN*5 f32) | S (B*H f32)  ~ 0.75 MB
  int*   nbr  = (int*)d_ws;
  float* sims = (float*)(nbr + (size_t)BN * KNN);
  float* Srow = sims + (size_t)BN * KNN;
  float* out  = (float*)d_out;

  topk_kernel<<<B, n, 2 * n * (int)sizeof(float), stream>>>(spatials, RoI_mask, nbr, n);
  rowsum_kernel<<<B, 256, 0, stream>>>(encI, Srow, n, H);
  sim_wmma_kernel<<<B * (n / 16), 32, 0, stream>>>(encI, nbr, sims, n, H, n / 16);
  output_kernel<<<BN, 192, 0, stream>>>(encI, nbr, sims, Srow, out, n, H);
}
